// Attention_56530359550448
// MI455X (gfx1250) — matrix operand order repaired and checked by reading, not compiled, not run on hardware
//
#include <hip/hip_runtime.h>
#include <hip/hip_bf16.h>
#include <math.h>

// ---------------------------------------------------------------------------
// Problem constants (from reference)
// ---------------------------------------------------------------------------
#define BB    4
#define NN    2048
#define DIM   768
#define HH    8
#define DH    96
#define INNER 768          // H*DH
#define QKVW  2304         // 3*INNER
#define ROWS  8192         // B*N
static constexpr float SCALE = 0.1020620726159657f;  // 96^-0.5

// ---------------------------------------------------------------------------
// CDNA5 WMMA / TDM types
// ---------------------------------------------------------------------------
typedef __attribute__((ext_vector_type(16))) __bf16        v16bf;
typedef __attribute__((ext_vector_type(8)))  float         v8f;
typedef __attribute__((ext_vector_type(4))) unsigned int  u32x4;
typedef __attribute__((ext_vector_type(8)))  int           i32x8;
typedef __attribute__((ext_vector_type(4)))  int           i32x4;

union FragCast { u32x4 q[2]; v16bf v; };

static __device__ inline v8f wmma_bf16(v16bf a, v16bf b, v8f c) {
    return __builtin_amdgcn_wmma_f32_16x16x32_bf16(
        /*neg_a=*/false, a, /*neg_b=*/false, b,
        /*c_mod=*/(short)0, c, /*reuse_a=*/false, /*reuse_b=*/false);
}

// A-matrix fragment (16x32 bf16, M-major, strideH in halves).  ISA 7.12.2:
// lanes 0-15 -> M=lane, K {0..7,16..23}; lanes 16-31 -> M=lane-16, K {8..15,24..31}.
static __device__ inline v16bf load_frag_a(const __bf16* base, int strideH) {
    const int lane = threadIdx.x & 31;
    const int r    = lane & 15;
    const int kb   = (lane < 16) ? 0 : 8;
    const __bf16* p = base + (size_t)r * strideH + kb;
    FragCast f;
    f.q[0] = *(const u32x4*)(p);
    f.q[1] = *(const u32x4*)(p + 16);
    return f.v;
}

// B-matrix fragment (32x16 bf16) read from an N-major (transposed) tile:
// lanes 0-15 hold K {0..7,16..23}, lanes 16-31 hold K {8..15,24..31} (matches A).
static __device__ inline v16bf load_frag_b(const __bf16* base, int strideH) {
    const int lane = threadIdx.x & 31;
    const int r    = lane & 15;
    const int kb   = (lane < 16) ? 0 : 8;
    const __bf16* p = base + (size_t)r * strideH + kb;
    FragCast f;
    f.q[0] = *(const u32x4*)(p);
    f.q[1] = *(const u32x4*)(p + 16);
    return f.v;
}

// ---------------------------------------------------------------------------
// Tensor Data Mover: 2D tile (rows x cols bf16, row pitch in elements) -> LDS.
// D# per ISA 8.3/8.4.  Group0: count=1, lds_addr, global_addr, type=2.
// Group1: data_size=2B, tensor dims == tile dims (no OOB), dim0 stride = pitch.
// This toolchain exposes the 6-arg builtin (groups 2/3 + extra group, + cpol).
// ---------------------------------------------------------------------------
static __device__ inline void tdm_load_tile(__bf16* lds_dst, const __bf16* gsrc,
                                            unsigned int cols, unsigned int rows,
                                            unsigned long long pitchElems) {
    const unsigned int       laddr = (unsigned int)(unsigned long long)(uintptr_t)lds_dst;
    const unsigned long long gaddr = (unsigned long long)(uintptr_t)gsrc;

    u32x4 g0;
    g0.x = 1u;                                            // count=1
    g0.y = laddr;                                         // lds_addr
    g0.z = (unsigned int)gaddr;                           // global_addr[31:0]
    g0.w = (unsigned int)(gaddr >> 32) | (2u << 30);      // global_addr[56:32] | type=2

    i32x8 g1;
    g1[0] = (int)(1u << 16);                              // data_size = 2 bytes
    g1[1] = (int)((cols & 0xFFFFu) << 16);                // tensor_dim0[15:0]
    g1[2] = (int)(((cols >> 16) & 0xFFFFu) | ((rows & 0xFFFFu) << 16));  // d0 hi | d1 lo
    g1[3] = (int)(((rows >> 16) & 0xFFFFu) | ((cols & 0xFFFFu) << 16));  // d1 hi | tile_dim0
    g1[4] = (int)(rows & 0xFFFFu);                        // tile_dim1 (tile_dim2 = 0)
    g1[5] = (int)(unsigned int)(pitchElems & 0xFFFFFFFFull);     // dim0_stride lo
    g1[6] = (int)(unsigned int)((pitchElems >> 32) & 0xFFFFull); // dim0_stride hi
    g1[7] = 0;

    const i32x4 z4 = {0, 0, 0, 0};
    const i32x8 z8 = {0, 0, 0, 0, 0, 0, 0, 0};
    __builtin_amdgcn_tensor_load_to_lds(g0, g1, z4, z4, z8, 0);
}

// ---------------------------------------------------------------------------
// fp32 -> bf16 conversion
// ---------------------------------------------------------------------------
__global__ void f32_to_bf16_kernel(const float* __restrict__ in,
                                   __bf16* __restrict__ out, int n) {
    int i = blockIdx.x * blockDim.x + threadIdx.x;
    if (i < n) out[i] = (__bf16)in[i];
}

// ---------------------------------------------------------------------------
// Generic linear:  C[M,Nn] = A[M,K] * W[Nn,K]^T  (+bias)
// Block = 256 thr (8 waves), tile 128x128, K step 32.
// TDM-fed double-buffered LDS: wave 0 issues tensor_load_to_lds for stage s+1,
// waits TENSORcnt<=2 (stage s landed), barrier, all waves run 8 WMMAs each.
// ---------------------------------------------------------------------------
template <bool OUTF32, bool BIAS>
__global__ __launch_bounds__(256)
void gemm_xwT(const __bf16* __restrict__ A, const __bf16* __restrict__ W,
              const float* __restrict__ bias, void* __restrict__ Cout,
              int M, int Nn, int K) {
    __shared__ __bf16 As[2][128 * 32];   // M-major
    __shared__ __bf16 Bs[2][128 * 32];   // N-major (W rows are already [n][k])

    const int tid  = threadIdx.x;
    const int wave = tid >> 5;
    const int lane = tid & 31;
    const int mBase = blockIdx.y * 128;
    const int nBase = blockIdx.x * 128;
    const int wm = (wave >> 1) * 32;
    const int wn = (wave & 1) * 64;

    const __bf16* Abase = A + (size_t)mBase * K;
    const __bf16* Wbase = W + (size_t)nBase * K;

    v8f acc[2][4];
    const v8f vzero = {0.f, 0.f, 0.f, 0.f, 0.f, 0.f, 0.f, 0.f};
#pragma unroll
    for (int i = 0; i < 2; ++i)
#pragma unroll
        for (int j = 0; j < 4; ++j) acc[i][j] = vzero;

    const int nsteps = K >> 5;
    // Prologue: stage 0 in flight.
    if (wave == 0) {
        tdm_load_tile(As[0], Abase, 32, 128, (unsigned long long)K);
        tdm_load_tile(Bs[0], Wbase, 32, 128, (unsigned long long)K);
    }

    for (int s = 0; s < nsteps; ++s) {
        const int cur = s & 1;
        __syncthreads();                      // buffer cur^1 fully consumed
        if (wave == 0) {
            if (s + 1 < nsteps) {
                tdm_load_tile(As[cur ^ 1], Abase + (size_t)(s + 1) * 32, 32, 128,
                              (unsigned long long)K);
                tdm_load_tile(Bs[cur ^ 1], Wbase + (size_t)(s + 1) * 32, 32, 128,
                              (unsigned long long)K);
                __builtin_amdgcn_s_wait_tensorcnt((short)2);   // stage s landed
            } else {
                __builtin_amdgcn_s_wait_tensorcnt((short)0);
            }
        }
        __syncthreads();                      // buffer cur ready for everyone

        v16bf af[2], bfr[4];
#pragma unroll
        for (int i = 0; i < 2; ++i)
            af[i] = load_frag_a(&As[cur][(wm + i * 16) * 32], 32);
#pragma unroll
        for (int j = 0; j < 4; ++j)
            bfr[j] = load_frag_b(&Bs[cur][(wn + j * 16) * 32], 32);
#pragma unroll
        for (int i = 0; i < 2; ++i)
#pragma unroll
            for (int j = 0; j < 4; ++j)
                acc[i][j] = wmma_bf16(af[i], bfr[j], acc[i][j]);
    }

    // C/D layout: VGPR r -> M = r + 8*(lane>=16); N = lane&15.
    const int hi = lane >> 4;
    const int cl = lane & 15;
#pragma unroll
    for (int i = 0; i < 2; ++i)
#pragma unroll
        for (int j = 0; j < 4; ++j)
#pragma unroll
            for (int r = 0; r < 8; ++r) {
                const int m = mBase + wm + i * 16 + r + 8 * hi;
                const int n = nBase + wn + j * 16 + cl;
                float v = acc[i][j][r];
                if (BIAS) v += bias[n];
                if (OUTF32)
                    ((float*)Cout)[(size_t)m * Nn + n] = v;
                else
                    ((__bf16*)Cout)[(size_t)m * Nn + n] = (__bf16)v;
            }
}

// ---------------------------------------------------------------------------
// Flash attention.  qkv: bf16 [ROWS][2304] (Q|K|V, head h at col h*96).
// Block = 256 thr (8 waves); blockIdx.y = b*H+h, blockIdx.x = 128-query tile.
// K tile (32x96) arrives via TDM; V tile is manually transposed into LDS
// (TDM has no transpose); P bounces through per-wave LDS for C->A relayout.
// ---------------------------------------------------------------------------
__global__ __launch_bounds__(256)
void flash_attn_kernel(const __bf16* __restrict__ qkv,
                       __bf16* __restrict__ obuf) {
    __shared__ __bf16 Klds[32 * 96];       // [key j][d]
    __shared__ __bf16 Vt[96 * 32];         // [d][key j]
    __shared__ __bf16 Plds[8 * 16 * 32];   // per-wave P tile, row-major 16x32

    const int tid  = threadIdx.x;
    const int wave = tid >> 5;
    const int lane = tid & 31;
    const int bh = blockIdx.y;
    const int b  = bh >> 3;
    const int h  = bh & 7;
    const int qbase = blockIdx.x * 128 + wave * 16;

    // Q fragments resident in registers, A-layout (16 rows x 96 K).
    const __bf16* qptr = qkv + (size_t)(b * NN + qbase) * QKVW + h * DH;
    v16bf qf[3];
#pragma unroll
    for (int d = 0; d < 3; ++d) qf[d] = load_frag_a(qptr + d * 32, QKVW);

    float rmax[8], rsum[8];
    v8f acc[6];
    const v8f vzero = {0.f, 0.f, 0.f, 0.f, 0.f, 0.f, 0.f, 0.f};
#pragma unroll
    for (int r = 0; r < 8; ++r) { rmax[r] = -1e30f; rsum[r] = 0.f; }
#pragma unroll
    for (int t = 0; t < 6; ++t) acc[t] = vzero;

    const int hi = lane >> 4;
    const int cl = lane & 15;
    __bf16* pl = &Plds[wave * (16 * 32)];

    for (int kb = 0; kb < NN / 32; ++kb) {
        __syncthreads();
        const __bf16* kg = qkv + (size_t)(b * NN + kb * 32) * QKVW + INNER + h * DH;
        const __bf16* vg = kg + INNER;

        // K tile via Tensor Data Mover (strided 2D: 32 rows x 96 halves).
        if (wave == 0)
            tdm_load_tile(Klds, kg, DH, 32, (unsigned long long)QKVW);

        // V tile: manual transposed staging (runs concurrently with the TDM).
        for (int c = tid; c < 384; c += 256) {
            const int j  = c / 12;
            const int dc = (c % 12) * 8;
            u32x4 vd = *(const u32x4*)&vg[(size_t)j * QKVW + dc];
            const __bf16* vh = (const __bf16*)&vd;
#pragma unroll
            for (int t = 0; t < 8; ++t) Vt[(dc + t) * 32 + j] = vh[t];
        }
        if (wave == 0) __builtin_amdgcn_s_wait_tensorcnt((short)0);
        __syncthreads();

        // Hoist all 6 K fragments, then 6 WMMAs (s0/s1 chains independent).
        v16bf kf[6];
#pragma unroll
        for (int d = 0; d < 3; ++d) {
            kf[2 * d]     = load_frag_b(&Klds[0 * 96 + d * 32], 96);
            kf[2 * d + 1] = load_frag_b(&Klds[16 * 96 + d * 32], 96);
        }
        v8f s0 = vzero, s1 = vzero;
#pragma unroll
        for (int d = 0; d < 3; ++d) {
            s0 = wmma_bf16(qf[d], kf[2 * d], s0);
            s1 = wmma_bf16(qf[d], kf[2 * d + 1], s1);
        }

        // Online softmax per row (row M=r+8*hi lives in one 16-lane group).
#pragma unroll
        for (int r = 0; r < 8; ++r) {
            float a  = s0[r] * SCALE;
            float bv = s1[r] * SCALE;
            float mx = fmaxf(a, bv);
            mx = fmaxf(mx, __shfl_xor(mx, 1, 32));
            mx = fmaxf(mx, __shfl_xor(mx, 2, 32));
            mx = fmaxf(mx, __shfl_xor(mx, 4, 32));
            mx = fmaxf(mx, __shfl_xor(mx, 8, 32));
            const float mnew  = fmaxf(rmax[r], mx);
            const float alpha = __expf(rmax[r] - mnew);
            rmax[r] = mnew;
            const float p0 = __expf(a - mnew);
            const float p1 = __expf(bv - mnew);
            float ps = p0 + p1;
            ps += __shfl_xor(ps, 1, 32);
            ps += __shfl_xor(ps, 2, 32);
            ps += __shfl_xor(ps, 4, 32);
            ps += __shfl_xor(ps, 8, 32);
            rsum[r] = rsum[r] * alpha + ps;
#pragma unroll
            for (int t = 0; t < 6; ++t) acc[t][r] *= alpha;
            const int prow = r + 8 * hi;
            pl[prow * 32 + cl]      = (__bf16)p0;
            pl[prow * 32 + 16 + cl] = (__bf16)p1;
        }
        // Wave-private LDS RAW: drain DS ops before re-reading as A-fragment.
        asm volatile("s_wait_dscnt 0" ::: "memory");

        v16bf pf = load_frag_a(pl, 32);
        v16bf vf[6];
#pragma unroll
        for (int t = 0; t < 6; ++t)
            vf[t] = load_frag_b(&Vt[(t * 16) * 32], 32);
#pragma unroll
        for (int t = 0; t < 6; ++t)
            acc[t] = wmma_bf16(pf, vf[t], acc[t]);
    }

    // O = acc / rowsum -> obuf[B*N][768] bf16 (merged heads).
    float inv[8];
#pragma unroll
    for (int r = 0; r < 8; ++r) inv[r] = 1.0f / rsum[r];
#pragma unroll
    for (int t = 0; t < 6; ++t)
#pragma unroll
        for (int r = 0; r < 8; ++r) {
            const int row = b * NN + qbase + r + 8 * hi;
            const int col = h * DH + t * 16 + cl;
            obuf[(size_t)row * INNER + col] = (__bf16)(acc[t][r] * inv[r]);
        }
}

// ---------------------------------------------------------------------------
// Host-side launcher
// ---------------------------------------------------------------------------
extern "C" void kernel_launch(void* const* d_in, const int* in_sizes, int n_in,
                              void* d_out, int out_size, void* d_ws, size_t ws_size,
                              hipStream_t stream) {
    const float* x      = (const float*)d_in[0];   // [B,N,DIM]
    const float* w_qkv  = (const float*)d_in[1];   // [2304, 768]
    const float* w_proj = (const float*)d_in[2];   // [768, 768]
    const float* b_proj = (const float*)d_in[3];   // [768]
    float* out = (float*)d_out;                    // [B,N,DIM]

    char* ws = (char*)d_ws;
    const size_t szXbf   = (size_t)ROWS * DIM * 2;
    const size_t szWqkv  = (size_t)QKVW * DIM * 2;
    const size_t szWproj = (size_t)DIM * INNER * 2;
    const size_t szQKV   = (size_t)ROWS * QKVW * 2;
    __bf16* Xbf    = (__bf16*)(ws);
    __bf16* Wqkvbf = (__bf16*)(ws + szXbf);
    __bf16* Wprojbf= (__bf16*)(ws + szXbf + szWqkv);
    __bf16* QKVbf  = (__bf16*)(ws + szXbf + szWqkv + szWproj);
    __bf16* Obf    = (__bf16*)(ws + szXbf + szWqkv + szWproj + szQKV);

    // 1) fp32 -> bf16 conversions
    {
        const int nx = ROWS * DIM;
        const int nq = QKVW * DIM;
        const int np = DIM * INNER;
        f32_to_bf16_kernel<<<(nx + 255) / 256, 256, 0, stream>>>(x, Xbf, nx);
        f32_to_bf16_kernel<<<(nq + 255) / 256, 256, 0, stream>>>(w_qkv, Wqkvbf, nq);
        f32_to_bf16_kernel<<<(np + 255) / 256, 256, 0, stream>>>(w_proj, Wprojbf, np);
    }

    // 2) QKV projection: [8192,2304] = X[8192,768] * Wqkv[2304,768]^T (bf16 out)
    gemm_xwT<false, false><<<dim3(QKVW / 128, ROWS / 128), 256, 0, stream>>>(
        Xbf, Wqkvbf, nullptr, (void*)QKVbf, ROWS, QKVW, DIM);

    // 3) Flash attention -> merged O [8192,768] bf16
    flash_attn_kernel<<<dim3(NN / 128, BB * HH), 256, 0, stream>>>(QKVbf, Obf);

    // 4) Output projection: out[8192,768] = O * Wproj^T + b (fp32 out)
    gemm_xwT<true, true><<<dim3(DIM / 128, ROWS / 128), 256, 0, stream>>>(
        Obf, Wprojbf, b_proj, (void*)out, ROWS, DIM, INNER);
}
